// QPLEXMixer_4329327034502
// MI455X (gfx1250) — matrix operand order repaired and checked by reading, not compiled, not run on hardware
//
#include <hip/hip_runtime.h>
#include <math.h>

// ---------------------------------------------------------------------------
// QPLEX / Qatten mixer fused kernel for gfx1250 (MI455X), wave32 + WMMA f16.
// Block = 32 batch rows (two 16-row WMMA M-tiles sharing every B-fragment),
// 8 wave32s. All inter-layer activations stay in LDS as f16.
// ---------------------------------------------------------------------------

#define B_TOTAL   131072
#define N_AGENTS  8
#define STATE_DIM 256
#define ACT_DIM   112
#define XDIM      384           // 368 padded to 384 (multiple of 32)
#define NKERN     10
#define H1W       640           // K*ADVH
#define TILE_B    32

typedef __attribute__((ext_vector_type(16))) _Float16 v16h;
typedef __attribute__((ext_vector_type(8)))  float    v8f;

// ---- f16 weight workspace layout (offsets in halves) ----------------------
#define OFF_SEL_W1 0                        // [256][256]
#define OFF_SEL_W2 (OFF_SEL_W1 + 65536)     // [128][64]
#define OFF_KEY_T  (OFF_SEL_W2 + 8192)      // [4][32(u)][32(e)] transposed
#define OFF_WH_W1  (OFF_KEY_T  + 4096)      // [64][256]
#define OFF_V_W1   (OFF_WH_W1  + 16384)     // [32][256]
#define OFF_WH_W2  (OFF_V_W1   + 8192)      // [4][64]
#define OFF_V_W2   (OFF_WH_W2  + 256)       // [32]
#define OFF_SIK_W1 (OFF_V_W2   + 32)        // [640][256]
#define OFF_SIK_W2 (OFF_SIK_W1 + 163840)    // [640][64]
#define OFF_SIK_W3 (OFF_SIK_W2 + 40960)     // [10][64]
#define OFF_SIA_W1 (OFF_SIK_W3 + 640)       // [640][256]
#define OFF_SIA_W2 (OFF_SIA_W1 + 163840)    // [640][64]
#define OFF_SIA_W3 (OFF_SIA_W2 + 40960)     // [10][16][64] agent-padded
#define OFF_SIC_W1 (OFF_SIA_W3 + 10240)     // [640][384] zero-padded K
#define OFF_SIC_W2 (OFF_SIC_W1 + 245760)    // [640][64]
#define OFF_SIC_W3 (OFF_SIC_W2 + 40960)     // [10][16][64] agent-padded
// total = 820128 halves = 1.64 MB of workspace

// ---------------------------------------------------------------------------
// WMMA fragment helpers (wave32; A and B both use contiguous k = 16*(lane/16) + i).
// ---------------------------------------------------------------------------
__device__ __forceinline__ v16h load_a_lds(const _Float16* A, int lda, int k0, int lane) {
  int row = lane & 15;
  const _Float16* p = A + row * lda + k0 + ((lane >> 4) << 4);
  v16h a;
#pragma unroll
  for (int i = 0; i < 8; ++i) a[i] = p[i];
#pragma unroll
  for (int i = 0; i < 8; ++i) a[8 + i] = p[8 + i];
  return a;
}

__device__ __forceinline__ v16h load_b_w(const _Float16* W, int ldb, int wn0, int k0, int lane) {
  const _Float16* p = W + (wn0 + (lane & 15)) * ldb + k0 + ((lane >> 4) << 4);
  v16h b;
#pragma unroll
  for (int i = 0; i < 16; ++i) b[i] = p[i];
  return b;
}

// Two stacked 16-row M-tiles against one B-fragment stream (raw accumulators).
__device__ __forceinline__ void gemm2_acc(const _Float16* A, int lda, int K,
                                          const _Float16* W, int ldb, int wn0,
                                          int lane, v8f& c0, v8f& c1) {
  c0 = (v8f){0.f, 0.f, 0.f, 0.f, 0.f, 0.f, 0.f, 0.f};
  c1 = (v8f){0.f, 0.f, 0.f, 0.f, 0.f, 0.f, 0.f, 0.f};
  for (int k0 = 0; k0 < K; k0 += 32) {
    v16h b  = load_b_w(W, ldb, wn0, k0, lane);
    v16h a0 = load_a_lds(A, lda, k0, lane);
    v16h a1 = load_a_lds(A + 16 * lda, lda, k0, lane);
    c0 = __builtin_amdgcn_wmma_f32_16x16x32_f16(false, a0, false, b, (short)0, c0,
                                                false, false);
    c1 = __builtin_amdgcn_wmma_f32_16x16x32_f16(false, a1, false, b, (short)0, c1,
                                                false, false);
  }
}

// 32x16 output tile: O = act(A[32xK] * W^T + bias), stored f16 in LDS.
__device__ __forceinline__ void gemm2_store(const _Float16* A, int lda, int K,
                                            const _Float16* W, int ldb, int wn0,
                                            const float* bias,
                                            _Float16* O, int ldo, int on0,
                                            int lane, bool relu) {
  v8f c0, c1;
  gemm2_acc(A, lda, K, W, ldb, wn0, lane, c0, c1);
  int col = lane & 15;
  int mb  = (lane >> 4) << 3;               // D: lanes 16-31 hold rows 8..15
  float bv = bias ? bias[wn0 + col] : 0.f;
  _Float16* o = O + on0 + col;
#pragma unroll
  for (int r = 0; r < 8; ++r) {
    float v0 = c0[r] + bv, v1 = c1[r] + bv;
    if (relu) { v0 = fmaxf(v0, 0.f); v1 = fmaxf(v1, 0.f); }
    o[(mb + r) * ldo]      = (_Float16)v0;
    o[(16 + mb + r) * ldo] = (_Float16)v1;
  }
}

// ---------------------------------------------------------------------------
// Prep kernels: f32 -> f16 weight conversion, transpose, pads, out zeroing.
// ---------------------------------------------------------------------------
__global__ void k_cvt(const float* __restrict__ s, _Float16* __restrict__ d, int n) {
  int i = blockIdx.x * 256 + threadIdx.x;
  if (i < n) d[i] = (_Float16)s[i];
}

__global__ void k_cvt_keyT(const float* __restrict__ s, _Float16* __restrict__ d) {
  int i = blockIdx.x * 256 + threadIdx.x;       // 4096 elements
  if (i < 4096) {
    int h = i >> 10, r = i & 1023, u = r >> 5, e = r & 31;
    d[i] = (_Float16)s[h * 1024 + e * 32 + u];  // [h][e][u] -> [h][u][e]
  }
}

__global__ void k_cvt_pad(const float* __restrict__ s, _Float16* __restrict__ d) {
  int i = blockIdx.x * 256 + threadIdx.x;       // 640*384
  if (i < 640 * 384) {
    int row = i / 384, col = i % 384;
    d[i] = (col < 368) ? (_Float16)s[row * 368 + col] : (_Float16)0.f;
  }
}

// [10][8][64] -> [10][16][64], agent rows 8..15 zero (clean WMMA N-tiles).
__global__ void k_cvt_padw3(const float* __restrict__ s, _Float16* __restrict__ d) {
  int i = blockIdx.x * 256 + threadIdx.x;       // 10240
  if (i < 10240) {
    int j = i & 63, ka = i >> 6, k = ka >> 4, a = ka & 15;
    d[i] = (a < 8) ? (_Float16)s[(k * 8 + a) * 64 + j] : (_Float16)0.f;
  }
}

__global__ void k_zero_tail(float* o) {
  if (threadIdx.x < 5) o[B_TOTAL + threadIdx.x] = 0.f;
}

// ---------------------------------------------------------------------------
// Fused mixer kernel.
// ---------------------------------------------------------------------------
__global__ __launch_bounds__(256) void qplex_main(
    const float* __restrict__ agent_qs, const float* __restrict__ states,
    const float* __restrict__ actions,  const float* __restrict__ max_q,
    const float* __restrict__ sel_b1, const float* __restrict__ wh_b1,
    const float* __restrict__ wh_b2,  const float* __restrict__ v_b1,
    const float* __restrict__ v_b2,
    const float* __restrict__ sik_b1, const float* __restrict__ sik_b2,
    const float* __restrict__ sik_b3,
    const float* __restrict__ sia_b1, const float* __restrict__ sia_b2,
    const float* __restrict__ sia_b3,
    const float* __restrict__ sic_b1, const float* __restrict__ sic_b2,
    const float* __restrict__ sic_b3,
    const _Float16* __restrict__ ws, float* __restrict__ out) {
  __shared__ _Float16 sX[TILE_B * XDIM];   // states | actions | pad (f16)
  __shared__ _Float16 sH1[TILE_B * H1W];   // layer-1 activations (reused)
  __shared__ _Float16 sH2[TILE_B * H1W];   // layer-2 activations (reused)
  __shared__ _Float16 sXag[TILE_B * NKERN * N_AGENTS];
  __shared__ float sLA[TILE_B * 32];       // logits -> attention weights
  __shared__ float sWH[TILE_B * 4];        // |w_head|
  __shared__ float sV[TILE_B];
  __shared__ float sQ[TILE_B * 8], sAQv[TILE_B * 8], sAW[TILE_B * 8];
  __shared__ float sXK[TILE_B * NKERN];    // x_key
  __shared__ float sAQ[TILE_B * 8], sMQ[TILE_B * 8];
  __shared__ float sEnt[4];
  __shared__ float sReg;

  const int tid = threadIdx.x, lane = tid & 31, wave = tid >> 5;
  const int b0 = blockIdx.x * TILE_B;

  // ---- Phase 0: stage inputs into LDS (f32 -> f16) ----
  for (int i = tid; i < TILE_B * STATE_DIM; i += 256) {
    int r = i >> 8, c = i & 255;
    sX[r * XDIM + c] = (_Float16)states[(b0 + r) * STATE_DIM + c];
  }
  for (int i = tid; i < TILE_B * ACT_DIM; i += 256) {
    int r = i / ACT_DIM, c = i % ACT_DIM;
    sX[r * XDIM + STATE_DIM + c] = (_Float16)actions[(b0 + r) * ACT_DIM + c];
  }
  for (int i = tid; i < TILE_B * 16; i += 256) {   // zero K-pad 368..383
    int r = i >> 4, c = i & 15;
    sX[r * XDIM + 368 + c] = (_Float16)0.f;
  }
  for (int i = tid; i < TILE_B * 8; i += 256) {
    sAQ[i] = agent_qs[b0 * 8 + i];
    sMQ[i] = max_q[b0 * 8 + i];
    sAW[i] = 0.f;
  }
  if (tid < 4) sEnt[tid] = 0.f;
  if (tid == 0) sReg = 0.f;
  __syncthreads();

  // ---- Phase 1: Qatten layer-1 GEMMs (sel 256 | wh 64 | v 32 cols) ----
  for (int t = wave; t < 22; t += 8) {
    int n0 = t * 16;
    const _Float16* W; const float* bias; int wn0;
    if (t < 16)      { W = ws + OFF_SEL_W1; bias = sel_b1; wn0 = n0;       }
    else if (t < 20) { W = ws + OFF_WH_W1;  bias = wh_b1;  wn0 = n0 - 256; }
    else             { W = ws + OFF_V_W1;   bias = v_b1;   wn0 = n0 - 320; }
    gemm2_store(sX, XDIM, STATE_DIM, W, 256, wn0, bias, sH1, H1W, n0, lane, true);
  }
  __syncthreads();

  // ---- Phase 2a: sel = H_sel @ sel_w2^T (per head, [32x32]) ----
  {
    int h = wave >> 1, nh = (wave & 1) * 16;
    gemm2_store(sH1 + h * 64, H1W, 64, ws + OFF_SEL_W2 + h * 32 * 64, 64, nh,
                nullptr, sH2, H1W, h * 32 + nh, lane, false);
  }
  __syncthreads();
  // ---- Phase 2b: t = sel @ key_w -> sH2 cols 128.. ----
  {
    int h = wave >> 1, nh = (wave & 1) * 16;
    gemm2_store(sH2 + h * 32, H1W, 32, ws + OFF_KEY_T + h * 32 * 32, 32, nh,
                nullptr, sH2, H1W, 128 + h * 32 + nh, lane, false);
  }
  __syncthreads();

  // ---- Phase 3: logits / softmax / V / w_head / w_final (VALU) ----
  for (int i = tid; i < TILE_B * 32; i += 256) {   // logits[b,h,a] = t . unit[a]
    int r = i >> 5, h = (i >> 3) & 3, a = i & 7;
    float s = 0.f;
    for (int u = 0; u < 32; ++u)
      s += (float)sH2[r * H1W + 128 + h * 32 + u] *
           (float)sX[r * XDIM + a * 32 + u];
    sLA[(r * 4 + h) * 8 + a] = s;
  }
  if (tid < TILE_B) {                              // V(s)
    float s = 0.f;
    for (int j = 0; j < 32; ++j)
      s += (float)sH1[tid * H1W + 320 + j] * (float)ws[OFF_V_W2 + j];
    sV[tid] = s + v_b2[0];
  }
  if (tid < TILE_B * 4) {                          // |w_head|
    int r = tid >> 2, h = tid & 3;
    float s = 0.f;
    for (int j = 0; j < 64; ++j)
      s += (float)sH1[r * H1W + 256 + j] * (float)ws[OFF_WH_W2 + h * 64 + j];
    sWH[tid] = fabsf(s + wh_b2[h]);
  }
  __syncthreads();

  if (tid < TILE_B * 4) {                          // softmax + entropy + reg
    float* L = &sLA[tid * 8];
    const float inv = 0.17677669529663687f;        // 1/sqrt(EMBED)
    float l[8], ssq = 0.f, mx = -1e30f;
    for (int a = 0; a < 8; ++a) {
      float x = L[a]; ssq += x * x;
      l[a] = x * inv; mx = fmaxf(mx, l[a]);
    }
    float sum = 0.f;
    for (int a = 0; a < 8; ++a) { l[a] = expf(l[a] - mx); sum += l[a]; }
    float rcp = 1.f / sum, ent = 0.f;
    for (int a = 0; a < 8; ++a) {
      float w = l[a] * rcp;
      L[a] = w;                                    // overwrite logits with attw
      ent -= w * logf(w + 1e-8f);
    }
    atomicAdd(&sReg, ssq);
    atomicAdd(&sEnt[tid & 3], ent);
  }
  __syncthreads();

  {                                                // w_final, q, adv_q (256 thr)
    int r = tid >> 3, a = tid & 7;
    float wf = 1e-10f;
    for (int h = 0; h < 4; ++h)
      wf += sLA[(r * 4 + h) * 8 + a] * sWH[r * 4 + h];
    float vper = sV[r] * (1.f / N_AGENTS);
    sQ[tid]   = wf * sAQ[tid] + vper;
    sAQv[tid] = wf * (sAQ[tid] - sMQ[tid]);        // q - mq
  }
  __syncthreads();

  // ---- Phase 4: the three K=10 hyper-MLPs ----
  for (int g = 0; g < 3; ++g) {
    const _Float16 *W1, *W2, *W3;
    const float *B1, *B2, *B3;
    int K1;
    if (g == 0) { W1 = ws + OFF_SIK_W1; W2 = ws + OFF_SIK_W2; W3 = ws + OFF_SIK_W3;
                  B1 = sik_b1; B2 = sik_b2; B3 = sik_b3; K1 = 256; }
    else if (g == 1) { W1 = ws + OFF_SIA_W1; W2 = ws + OFF_SIA_W2; W3 = ws + OFF_SIA_W3;
                  B1 = sia_b1; B2 = sia_b2; B3 = sia_b3; K1 = 256; }
    else { W1 = ws + OFF_SIC_W1; W2 = ws + OFF_SIC_W2; W3 = ws + OFF_SIC_W3;
                  B1 = sic_b1; B2 = sic_b2; B3 = sic_b3; K1 = 384; }
    // L1: [32x640] = relu(X @ W1^T + b1)
    for (int t = wave; t < 40; t += 8)
      gemm2_store(sX, XDIM, K1, W1, K1, t * 16, B1, sH1, H1W, t * 16, lane, true);
    __syncthreads();
    // L2: block-diagonal per kernel k, K = 64
    for (int t = wave; t < 40; t += 8) {
      int n0 = t * 16, k = n0 >> 6;
      gemm2_store(sH1 + k * 64, H1W, 64, W2, 64, n0, B2, sH2, H1W, n0, lane, true);
    }
    __syncthreads();
    // L3 output heads
    if (g == 0) {                                  // x_key: 10 scalars/row (VALU)
      for (int i = tid; i < TILE_B * NKERN; i += 256) {
        int r = i / NKERN, k = i % NKERN;
        float s = 0.f;
        for (int j = 0; j < 64; ++j)
          s += (float)sH2[r * H1W + k * 64 + j] * (float)W3[k * 64 + j];
        sXK[i] = fabsf(s + B3[k]) + 1e-10f;
      }
    } else {
      // x_ag / x_ac via WMMA on agent-padded W3: tile k = 16 cols (8 valid).
      for (int k = wave; k < NKERN; k += 8) {
        v8f c0, c1;
        gemm2_acc(sH2 + k * 64, H1W, 64, W3 + k * 16 * 64, 64, 0, lane, c0, c1);
        int a  = lane & 15;
        int mb = (lane >> 4) << 3;
        if (a < 8) {
          float bv = B3[k * 8 + a];
#pragma unroll
          for (int i = 0; i < 8; ++i) {
            int r0 = mb + i, r1 = 16 + mb + i;
            float sg0 = 1.f / (1.f + expf(-(c0[i] + bv)));
            float sg1 = 1.f / (1.f + expf(-(c1[i] + bv)));
            if (g == 1) {
              sXag[(r0 * NKERN + k) * 8 + a] = (_Float16)sg0;
              sXag[(r1 * NKERN + k) * 8 + a] = (_Float16)sg1;
            } else {
              atomicAdd(&sAW[r0 * 8 + a],
                        sXK[r0 * NKERN + k] * (float)sXag[(r0 * NKERN + k) * 8 + a] * sg0);
              atomicAdd(&sAW[r1 * 8 + a],
                        sXK[r1 * NKERN + k] * (float)sXag[(r1 * NKERN + k) * 8 + a] * sg1);
            }
          }
        }
      }
    }
    __syncthreads();
  }

  // ---- Phase 5: q_tot + global scalar reductions ----
  if (tid < TILE_B) {
    int r = tid;
    float qs = 0.f, at = 0.f;
    for (int a = 0; a < 8; ++a) {
      qs += sQ[r * 8 + a];
      at += sAQv[r * 8 + a] * (sAW[r * 8 + a] - 1.f);
    }
    out[b0 + r] = qs + at;
  }
  if (tid == 0)
    atomicAdd(&out[B_TOTAL], sReg * (1e-3f / ((float)B_TOTAL * 8.f)));
  if (tid < 4)
    atomicAdd(&out[B_TOTAL + 1 + tid], sEnt[tid] * (1.f / (float)B_TOTAL));
}

// ---------------------------------------------------------------------------
extern "C" void kernel_launch(void* const* d_in, const int* in_sizes, int n_in,
                              void* d_out, int out_size, void* d_ws, size_t ws_size,
                              hipStream_t stream) {
  const float* agent_qs = (const float*)d_in[0];
  const float* states   = (const float*)d_in[1];
  const float* actions  = (const float*)d_in[2];
  const float* max_q    = (const float*)d_in[3];
  // params dict order:
  // 0 sel_w1  1 sel_b1  2 sel_w2  3 key_w  4 wh_w1  5 wh_b1  6 wh_w2  7 wh_b2
  // 8 v_w1    9 v_b1   10 v_w2   11 v_b2  12 sik_w1 13 sik_b1 14 sik_w2 15 sik_b2
  // 16 sik_w3 17 sik_b3 18 sia_w1 19 sia_b1 20 sia_w2 21 sia_b2 22 sia_w3 23 sia_b3
  // 24 sic_w1 25 sic_b1 26 sic_w2 27 sic_b2 28 sic_w3 29 sic_b3
  const float* P[30];
  for (int i = 0; i < 30; ++i) P[i] = (const float*)d_in[4 + i];
  _Float16* ws = (_Float16*)d_ws;
  float* out = (float*)d_out;

  auto cvt = [&](const float* s, int off, int n) {
    k_cvt<<<(n + 255) / 256, 256, 0, stream>>>(s, ws + off, n);
  };
  cvt(P[0],  OFF_SEL_W1, 65536);
  cvt(P[2],  OFF_SEL_W2, 8192);
  k_cvt_keyT<<<16, 256, 0, stream>>>(P[3], ws + OFF_KEY_T);
  cvt(P[4],  OFF_WH_W1, 16384);
  cvt(P[6],  OFF_WH_W2, 256);
  cvt(P[8],  OFF_V_W1, 8192);
  cvt(P[10], OFF_V_W2, 32);
  cvt(P[12], OFF_SIK_W1, 163840);
  cvt(P[14], OFF_SIK_W2, 40960);
  cvt(P[16], OFF_SIK_W3, 640);
  cvt(P[18], OFF_SIA_W1, 163840);
  cvt(P[20], OFF_SIA_W2, 40960);
  k_cvt_padw3<<<40, 256, 0, stream>>>(P[22], ws + OFF_SIA_W3);
  k_cvt_pad<<<(640 * 384 + 255) / 256, 256, 0, stream>>>(P[24], ws + OFF_SIC_W1);
  cvt(P[26], OFF_SIC_W2, 40960);
  k_cvt_padw3<<<40, 256, 0, stream>>>(P[28], ws + OFF_SIC_W3);
  k_zero_tail<<<1, 32, 0, stream>>>(out);

  qplex_main<<<B_TOTAL / TILE_B, 256, 0, stream>>>(
      agent_qs, states, actions, max_q,
      P[1], P[5], P[7], P[9], P[11],
      P[13], P[15], P[17],
      P[19], P[21], P[23],
      P[25], P[27], P[29],
      ws, out);
}
